// Classifier_69818988363920
// MI455X (gfx1250) — matrix operand order repaired and checked by reading, not compiled, not run on hardware
//
#include <hip/hip_runtime.h>
#include <hip/hip_bf16.h>

// ---------------------------------------------------------------------------
// Fused: relu(x @ W1^T) -> mean-pool(J=24) -> @ W2^T
//   x  : [786432, 50] f32, W1 : [400, 50] f32, W2 : [10, 400] f32
//   out: [32768, 10] f32
//  * prep kernel converts W1 -> bf16 (padded [400][64]) ONCE into d_ws.
//  * main kernel DMAs the 51200 B bf16 W1 into LDS with the Tensor Data
//    Mover (tensor_load_to_lds + s_wait_tensorcnt, 6-arg clang-23 form),
//    overlapped with the per-block x->bf16 staging.
//  * GEMM on v_wmma_f32_16x16x32_bf16; fused ReLU + 24-row mean pooling;
//    final 2x10 logits per block fused via LDS.
// ---------------------------------------------------------------------------

typedef __attribute__((ext_vector_type(16))) __bf16 v16bf;
typedef __attribute__((ext_vector_type(8)))  __bf16 v8bf;
typedef __attribute__((ext_vector_type(8)))  float  v8f;
typedef __attribute__((ext_vector_type(4)))  unsigned int v4u;
typedef __attribute__((ext_vector_type(8)))  int v8i;
typedef __attribute__((ext_vector_type(4)))  int v4i;

union BFrag16 { v16bf v; v8bf h[2]; };

__device__ __forceinline__ __bf16 f2bf(float f) {
    unsigned u = __builtin_bit_cast(unsigned, f);
    u += 0x7FFFu + ((u >> 16) & 1u);            // round-to-nearest-even
    unsigned short s = (unsigned short)(u >> 16);
    return __builtin_bit_cast(__bf16, s);
}

#define T_TOTAL   786432
#define N_IN      50
#define K_FEAT    400
#define C_OUT     10
#define JWIN      24
#define ROWS_BLK  48                       // 3 M-tiles = 2 pooling windows
#define KPAD      64                       // N padded to 2 WMMA k-steps
#define NBLOCKS   (T_TOTAL / ROWS_BLK)     // 16384
#define W1_ELEMS  (K_FEAT * KPAD)          // 25600 bf16
#define W1_BYTES  (W1_ELEMS * 2)           // 51200 B
#define W1_DWORDS (W1_BYTES / 4)           // 12800

// ---- one-time W1 fp32 -> bf16 (zero-padded K) into workspace --------------
__global__ __launch_bounds__(256) void convert_w1_kernel(
        const float* __restrict__ W1, __bf16* __restrict__ w1bf)
{
    int idx = blockIdx.x * 256 + threadIdx.x;
    if (idx < W1_ELEMS) {
        int row = idx >> 6, n = idx & (KPAD - 1);
        float v = (n < N_IN) ? W1[row * N_IN + n] : 0.0f;
        w1bf[idx] = f2bf(v);
    }
}

__global__ __launch_bounds__(256) void fused_classifier_kernel(
        const float* __restrict__ x,
        const __bf16* __restrict__ w1bf,
        const float* __restrict__ W2,
        float* __restrict__ out)
{
    __shared__ __align__(16) __bf16 sW1[K_FEAT][KPAD];   // 51200 B (TDM dest)
    __shared__ __align__(16) __bf16 sX[ROWS_BLK][KPAD];  //  6144 B
    __shared__ float sF[2][K_FEAT];                      //  3200 B

    const int tid     = threadIdx.x;
    const int rowBase = blockIdx.x * ROWS_BLK;

#if __has_builtin(__builtin_amdgcn_tensor_load_to_lds)
    // ---- wave 0: issue one TDM descriptor: 12800 contiguous dwords -------
    if (tid < 32) {
        unsigned lds_off = (unsigned)(unsigned long long)(void*)&sW1[0][0];
        unsigned long long ga = (unsigned long long)w1bf;
        v4u g0;
        g0[0] = 1u;                                    // count=1, user mode
        g0[1] = lds_off;                               // D#.lds_addr
        g0[2] = (unsigned)ga;                          // global_addr[31:0]
        g0[3] = (unsigned)((ga >> 32) & 0x01FFFFFFull) // global_addr[56:32]
              | 0x80000000u;                           // type=2 ("image")
        v8i g1;
        g1[0] = 0x00020000;                 // wg_mask=0, data_size=2 (4B)
        g1[1] = (int)((W1_DWORDS & 0xFFFF) << 16);     // tensor_dim0 lo
        g1[2] = (int)(((W1_DWORDS >> 16) & 0xFFFF) | (1u << 16)); // dim0 hi | tensor_dim1=1
        g1[3] = (int)((W1_DWORDS & 0xFFFF) << 16);     // tile_dim0 = 12800
        g1[4] = 1;                                     // tile_dim1=1, tile_dim2=0
        g1[5] = W1_DWORDS;                             // tensor_dim0_stride lo
        g1[6] = (int)((W1_DWORDS & 0xFFFF) << 16);     // stride0 hi=0 | stride1 lo
        g1[7] = 0;                                     // stride1 hi
        v4i g2 = {0, 0, 0, 0};
        v4i g3 = {0, 0, 0, 0};
        v8i g4 = {0, 0, 0, 0, 0, 0, 0, 0};
        __builtin_amdgcn_tensor_load_to_lds(g0, g1, g2, g3, g4, 0);
    }
#endif

    // ---- all waves: stage this block's 48 rows of x -> bf16 LDS ----------
    for (int idx = tid; idx < ROWS_BLK * KPAD; idx += 256) {
        int row = idx >> 6, n = idx & (KPAD - 1);
        float v = (n < N_IN) ? x[(size_t)(rowBase + row) * N_IN + n] : 0.0f;
        sX[row][n] = f2bf(v);
    }

#if __has_builtin(__builtin_amdgcn_tensor_load_to_lds)
    if (tid < 32) __builtin_amdgcn_s_wait_tensorcnt(0);  // wave 0 owns TENSORcnt
#else
    // fallback: cooperative vectorized copy of pre-converted W1
    {
        const v8bf* src = (const v8bf*)w1bf;
        v8bf* dst = (v8bf*)&sW1[0][0];
        for (int idx = tid; idx < W1_ELEMS / 8; idx += 256) dst[idx] = src[idx];
    }
#endif
    __syncthreads();   // publishes sW1 (TDM) + sX to all waves

    // ---- WMMA GEMM + ReLU + fused 24-row mean pooling ---------------------
    const int wave = tid >> 5;
    const int lane = tid & 31;
    const int j    = lane & 15;   // B/D column within tile; A row M
    const int hi   = lane >> 4;   // lane half selects K sub-range (ISA map)

    for (int ct = wave; ct < K_FEAT / 16; ct += 8) {
        // B fragments: W1 rows (16*ct + j), two k-chunks, halves at n = 8*hi and 8*hi+16.
        const __bf16* brow = &sW1[16 * ct + j][0];
        BFrag16 b0, b1;
        b0.h[0] = *(const v8bf*)(brow + 8 * hi);
        b0.h[1] = *(const v8bf*)(brow + 8 * hi + 16);
        b1.h[0] = *(const v8bf*)(brow + 32 + 8 * hi);
        b1.h[1] = *(const v8bf*)(brow + 32 + 8 * hi + 16);

        float rtsum[3];                     // per-row-tile ReLU'd sums
#pragma unroll
        for (int rt = 0; rt < 3; ++rt) {
            const __bf16* arow = &sX[16 * rt + j][0];
            BFrag16 a0, a1;
            a0.h[0] = *(const v8bf*)(arow + 8 * hi);
            a0.h[1] = *(const v8bf*)(arow + 8 * hi + 16);
            a1.h[0] = *(const v8bf*)(arow + 32 + 8 * hi);
            a1.h[1] = *(const v8bf*)(arow + 32 + 8 * hi + 16);

            v8f acc = {};
            acc = __builtin_amdgcn_wmma_f32_16x16x32_bf16(
                      false, a0.v, false, b0.v, (short)0, acc, false, false);
            acc = __builtin_amdgcn_wmma_f32_16x16x32_bf16(
                      false, a1.v, false, b1.v, (short)0, acc, false, false);

            float t = 0.0f;
#pragma unroll
            for (int r = 0; r < 8; ++r) t += __builtin_fmaxf(acc[r], 0.0f);
            rtsum[rt] = t;
        }
        // rt=0 rows 0..15 -> win0; rt=2 rows 32..47 -> win1;
        // rt=1: lane half hi=0 holds rows 16..23 (win0), hi=1 rows 24..31 (win1)
        float p0 = rtsum[0] + (hi ? 0.0f : rtsum[1]);
        float p1 = rtsum[2] + (hi ? rtsum[1] : 0.0f);

        p0 += __shfl_xor(p0, 16, 32);   // fold the two lane-halves per column
        p1 += __shfl_xor(p1, 16, 32);
        if (hi == 0) {
            int col = 16 * ct + j;
            sF[0][col] = p0 * (1.0f / JWIN);
            sF[1][col] = p1 * (1.0f / JWIN);
        }
    }
    __syncthreads();

    // ---- final tiny GEMM: 2 pooled rows x W2^T -> 2x10 logits -------------
    if (tid < 160) {                 // waves 0..4 fully active (wave32)
        int o    = tid >> 3;         // 0..19
        int part = tid & 7;
        int s    = o / C_OUT;
        int c    = o % C_OUT;
        float acc = 0.0f;
        const float* w2row = W2 + c * K_FEAT;
#pragma unroll 5
        for (int k = part * 50; k < part * 50 + 50; ++k)
            acc += sF[s][k] * w2row[k];
        acc += __shfl_xor(acc, 1, 32);
        acc += __shfl_xor(acc, 2, 32);
        acc += __shfl_xor(acc, 4, 32);
        if (part == 0)
            out[(size_t)(blockIdx.x * 2 + s) * C_OUT + c] = acc;
    }
}

extern "C" void kernel_launch(void* const* d_in, const int* in_sizes, int n_in,
                              void* d_out, int out_size, void* d_ws, size_t ws_size,
                              hipStream_t stream) {
    (void)in_sizes; (void)n_in; (void)ws_size; (void)out_size;
    const float* x  = (const float*)d_in[0];
    const float* W1 = (const float*)d_in[1];
    const float* W2 = (const float*)d_in[2];
    float* out      = (float*)d_out;
    __bf16* w1bf    = (__bf16*)d_ws;

    convert_w1_kernel<<<(W1_ELEMS + 255) / 256, 256, 0, stream>>>(W1, w1bf);
    fused_classifier_kernel<<<NBLOCKS, 256, 0, stream>>>(x, w1bf, W2, out);
}
